// VQSign_28278064677385
// MI455X (gfx1250) — matrix operand order repaired and checked by reading, not compiled, not run on hardware
//
#include <hip/hip_runtime.h>
#include <hip/hip_bf16.h>

typedef __bf16 bf16;
typedef __attribute__((ext_vector_type(16))) __bf16 v16bf;
typedef __attribute__((ext_vector_type(8)))  __bf16 v8bf;
typedef __attribute__((ext_vector_type(8)))  float  v8f;

#define NB    16
#define NL    512
#define DD    512
#define NK    8192
#define NROWS (NB * NL)      // 8192 flat rows
#define N3    (3 * DD)       // 1536
#define SPLITN 8
#define PSTRIDE 520          // LDS panel row stride (bf16): 512 + 8 pad -> conflict-free b128 reads
#define PANEL32 (32 * PSTRIDE)   // one 32-row stage (elems)

// sched_group_barrier instruction-class masks
#define SCHED_WMMA   0x008   // MFMA/WMMA
#define SCHED_VMREAD 0x020   // VMEM read
#define SCHED_DSREAD 0x100   // DS read

union FragU { v16bf v; v8bf h[2]; };

// ---- WMMA fragment loaders (contiguous k order shared by A and B, wave32) -----------
// A (16x32 bf16, M x K): lane m=lane&15 holds K = kb..kb+15 contiguous,
// kb = kofs + (lane>>4)*16. One contiguous 32B chunk per lane.
static __device__ inline v16bf load_frag_a(const bf16* arow, int hi, int kofs) {
    FragU u;
    u.h[0] = *(const v8bf*)(arow + kofs + hi * 16);
    u.h[1] = *(const v8bf*)(arow + kofs + hi * 16 + 8);
    return u.v;
}
// B (32x16 bf16, K x N): lane n=lane&15 holds K = kb..kb+15 contiguous,
// kb = kofs + (lane>>4)*16. One contiguous 32B chunk per lane.
static __device__ inline v16bf load_frag_b(const bf16* brow_khi, int kofs) {
    FragU u;
    u.h[0] = *(const v8bf*)(brow_khi + kofs);
    u.h[1] = *(const v8bf*)(brow_khi + kofs + 8);
    return u.v;
}

// K=512 dual-tile MAC. LOADMASK selects the instruction class of the B-frag
// loads (DS read for LDS-staged panels, VMEM read for global streaming).
// sched_group_barriers pin a software pipeline: 4 frags (8 loads) of lookahead,
// then one frag load per WMMA, so the allocator must keep multiple frag
// buffers live and the waitcnt inserter emits partial waits.
template<int LOADMASK>
static __device__ inline void k512_mac2_t(const v16bf* afr,
                                          const bf16* brow0, const bf16* brow1,
                                          v8f& acc0, v8f& acc1) {
    #pragma unroll
    for (int kc = 0; kc < 16; kc += 2) {
        v16bf b00 = load_frag_b(brow0, kc * 32);
        v16bf b01 = load_frag_b(brow1, kc * 32);
        v16bf b10 = load_frag_b(brow0, (kc + 1) * 32);
        v16bf b11 = load_frag_b(brow1, (kc + 1) * 32);
        acc0 = __builtin_amdgcn_wmma_f32_16x16x32_bf16(
                   false, afr[kc], false, b00, (short)0, acc0, false, false);
        acc1 = __builtin_amdgcn_wmma_f32_16x16x32_bf16(
                   false, afr[kc], false, b01, (short)0, acc1, false, false);
        acc0 = __builtin_amdgcn_wmma_f32_16x16x32_bf16(
                   false, afr[kc + 1], false, b10, (short)0, acc0, false, false);
        acc1 = __builtin_amdgcn_wmma_f32_16x16x32_bf16(
                   false, afr[kc + 1], false, b11, (short)0, acc1, false, false);
    }
    // Pipeline shape for the 64 loads + 32 WMMAs above:
    // 8 loads ahead, then [2 loads, 1 WMMA] x28, then drain 4 WMMAs.
    __builtin_amdgcn_sched_group_barrier(LOADMASK, 8, 0);
    #pragma unroll
    for (int i = 0; i < 28; ++i) {
        __builtin_amdgcn_sched_group_barrier(LOADMASK, 2, 0);
        __builtin_amdgcn_sched_group_barrier(SCHED_WMMA, 1, 0);
    }
    __builtin_amdgcn_sched_group_barrier(SCHED_WMMA, 4, 0);
}

// ---- CDNA5 async copy: global -> LDS, tracked by ASYNCcnt -------------------
// Copies a contiguous 32KB global panel (32 rows x 512 bf16) into an LDS panel
// with PSTRIDE row padding. 256 threads x 8 b128 chunks each.
static __device__ inline void issue_panel32(const bf16* gsrc, bf16* lpanel, int tid) {
    #pragma unroll
    for (int c = 0; c < 8; ++c) {
        int chunk = c * 256 + tid;          // 2048 chunks of 16B
        int row = chunk >> 6;               // 64 chunks per 1024B row
        int col = chunk & 63;
        unsigned lds = (unsigned)(unsigned long long)(const void*)
                       ((const char*)lpanel + row * (PSTRIDE * 2) + col * 16);
        unsigned long long g = (unsigned long long)((const char*)gsrc + row * 1024 + col * 16);
        asm volatile("global_load_async_to_lds_b128 %0, %1, off"
                     :: "v"(lds), "v"(g) : "memory");
    }
}
static __device__ inline void wait_async_le8() { asm volatile("s_wait_asynccnt 0x8" ::: "memory"); }
static __device__ inline void wait_async_le0() { asm volatile("s_wait_asynccnt 0x0" ::: "memory"); }

static __device__ inline float logsig(float x) {
    float m = fminf(x, 0.0f);
    return m - log1pf(__expf(-fabsf(x)));
}

// ---- small utility kernels --------------------------------------------------
__global__ void kZero(float* p, int n) {
    int i = blockIdx.x * blockDim.x + threadIdx.x;
    if (i < n) p[i] = 0.0f;
}

__global__ void kConvert(const float* __restrict__ src, bf16* __restrict__ dst, long n) {
    long i = (long)blockIdx.x * blockDim.x + threadIdx.x;
    long stride = (long)gridDim.x * blockDim.x;
    for (; i < n; i += stride) dst[i] = (bf16)src[i];
}

// one wave per row: ||row||^2
__global__ void kRowNorm(const float* __restrict__ src, float* __restrict__ out, int rows) {
    int wave = threadIdx.x >> 5, lane = threadIdx.x & 31;
    int row = blockIdx.x * (blockDim.x >> 5) + wave;
    if (row >= rows) return;
    const float* p = src + (size_t)row * DD;
    float s = 0.0f;
    for (int j = lane; j < DD; j += 32) { float v = p[j]; s += v * v; }
    for (int m = 16; m >= 1; m >>= 1) s += __shfl_xor(s, m, 32);
    if (lane == 0) out[row] = s;
}

// ---- VQ nearest-neighbor: bf16 WMMA GEMM + per-row running argmin ----------
// grid (64, SPLITN), 256 threads (8 waves). Wave owns a 16-row A stripe in
// VGPRs and computes TWO 16x16 tiles per stage; 32-row codebook panels are
// double-buffered in LDS via async copies and shared by all 8 waves.
__global__ __launch_bounds__(256)
void kArgminGemm(const bf16* __restrict__ feat, const bf16* __restrict__ cb,
                 const float* __restrict__ cbn,
                 float* __restrict__ partDist, int* __restrict__ partIdx) {
    extern __shared__ bf16 panel[];                  // [2][PANEL32]
    int tid = threadIdx.x;
    int lane = tid & 31, wave = tid >> 5;
    int lo = lane & 15, hi = lane >> 4;
    int mbase = blockIdx.x * 128 + wave * 16;
    int nstart = blockIdx.y * (NK / SPLITN);
    const int NSTAGE = (NK / SPLITN) / 32;           // 32 stages of 32 rows

    issue_panel32(cb + (size_t)nstart * DD, panel, tid);    // prefetch stage 0 first

    // A stripe: 16 rows x 512 K, held in registers, reused for 1024 N columns
    v16bf afr[16];
    const bf16* arow = feat + (size_t)(mbase + lo) * DD;
    #pragma unroll
    for (int kc = 0; kc < 16; ++kc) afr[kc] = load_frag_a(arow, hi, kc * 32);

    float bmin[8];
    int   bidx[8];
    #pragma unroll
    for (int r = 0; r < 8; ++r) { bmin[r] = 3.4e38f; bidx[r] = 0; }

    for (int st = 0; st < NSTAGE; ++st) {
        bf16* cur = panel + (st & 1) * PANEL32;
        if (st + 1 < NSTAGE) {
            issue_panel32(cb + (size_t)(nstart + (st + 1) * 32) * DD,
                          panel + ((st & 1) ^ 1) * PANEL32, tid);
            wait_async_le8();                        // current stage complete, next in flight
        } else {
            wait_async_le0();
        }
        __syncthreads();                             // publish LDS writes to all waves

        const bf16* brow0 = cur + lo * PSTRIDE + hi * 16;
        const bf16* brow1 = brow0 + 16 * PSTRIDE;
        v8f acc0 = {}, acc1 = {};
        k512_mac2_t<SCHED_DSREAD>(afr, brow0, brow1, acc0, acc1);

        int nb0 = nstart + st * 32;
        float cn0 = cbn[nb0 + lo];
        float cn1 = cbn[nb0 + 16 + lo];
        #pragma unroll
        for (int r = 0; r < 8; ++r) {
            float d0 = cn0 - 2.0f * acc0[r];         // ||f||^2 constant per row
            float d1 = cn1 - 2.0f * acc1[r];
            if (d0 < bmin[r]) { bmin[r] = d0; bidx[r] = nb0 + lo; }
            if (d1 < bmin[r]) { bmin[r] = d1; bidx[r] = nb0 + 16 + lo; }
        }
        __syncthreads();                             // all reads done before next overwrite
    }
    // min-reduce across the 16 lanes of each half (same M set, different N)
    #pragma unroll
    for (int m = 1; m < 16; m <<= 1) {
        #pragma unroll
        for (int r = 0; r < 8; ++r) {
            float od = __shfl_xor(bmin[r], m, 32);
            int   oi = __shfl_xor(bidx[r], m, 32);
            if (od < bmin[r] || (od == bmin[r] && oi < bidx[r])) { bmin[r] = od; bidx[r] = oi; }
        }
    }
    if (lo == 0) {
        #pragma unroll
        for (int r = 0; r < 8; ++r) {
            int row = mbase + hi * 8 + r;
            partDist[(size_t)row * SPLITN + blockIdx.y] = bmin[r];
            partIdx [(size_t)row * SPLITN + blockIdx.y] = bidx[r];
        }
    }
}

__global__ void kCombine(const float* __restrict__ partDist, const int* __restrict__ partIdx,
                         int* __restrict__ idxOut, float* __restrict__ tokOut) {
    int row = blockIdx.x * blockDim.x + threadIdx.x;
    if (row >= NROWS) return;
    float best = partDist[(size_t)row * SPLITN];
    int   bi   = partIdx [(size_t)row * SPLITN];
    for (int s = 1; s < SPLITN; ++s) {
        float d = partDist[(size_t)row * SPLITN + s];
        int   i = partIdx [(size_t)row * SPLITN + s];
        if (d < best || (d == best && i < bi)) { best = d; bi = i; }
    }
    idxOut[row] = bi;
    tokOut[row] = (float)bi;
}

// gather quantized (f32 -> d_out, bf16 -> ws) + sum (f - q)^2
__global__ void kGather(const float* __restrict__ feat, const float* __restrict__ cb,
                        const int* __restrict__ idx,
                        float* __restrict__ qOut, bf16* __restrict__ qbf,
                        float* __restrict__ ssePtr) {
    int row = blockIdx.x;
    const float* c = cb + (size_t)idx[row] * DD;
    const float* f = feat + (size_t)row * DD;
    float s = 0.0f;
    for (int j = threadIdx.x; j < DD; j += blockDim.x) {
        float q = c[j];
        qOut[(size_t)row * DD + j] = q;
        qbf [(size_t)row * DD + j] = (bf16)q;
        float d = f[j] - q;
        s += d * d;
    }
    for (int m = 16; m >= 1; m >>= 1) s += __shfl_xor(s, m, 32);
    if ((threadIdx.x & 31) == 0) atomicAdd(ssePtr, s);
}

// ---- generic GEMM: out[M x N] = A[M x 512] * Bm[N x 512]^T + bias ----------
// grid.x: M blocks of 128 (8 waves x 16 rows); grid.y * nstagesPerY covers
// N in stages of 32 columns (two 16x16 tiles per wave per stage).
__global__ __launch_bounds__(256)
void kGemmBias(const bf16* __restrict__ A, const bf16* __restrict__ Bm,
               const float* __restrict__ bias, float* __restrict__ out,
               int N, int nstagesPerY) {
    extern __shared__ bf16 panel[];                  // [2][PANEL32]
    int tid = threadIdx.x;
    int lane = tid & 31, wave = tid >> 5;
    int lo = lane & 15, hi = lane >> 4;
    int mbase = blockIdx.x * 128 + wave * 16;
    int stage0 = blockIdx.y * nstagesPerY;

    issue_panel32(Bm + (size_t)stage0 * 32 * DD, panel, tid);

    v16bf afr[16];
    const bf16* arow = A + (size_t)(mbase + lo) * DD;
    #pragma unroll
    for (int kc = 0; kc < 16; ++kc) afr[kc] = load_frag_a(arow, hi, kc * 32);

    for (int st = 0; st < nstagesPerY; ++st) {
        bf16* cur = panel + (st & 1) * PANEL32;
        if (st + 1 < nstagesPerY) {
            issue_panel32(Bm + (size_t)(stage0 + st + 1) * 32 * DD,
                          panel + ((st & 1) ^ 1) * PANEL32, tid);
            wait_async_le8();
        } else {
            wait_async_le0();
        }
        __syncthreads();

        const bf16* brow0 = cur + lo * PSTRIDE + hi * 16;
        const bf16* brow1 = brow0 + 16 * PSTRIDE;
        v8f acc0 = {}, acc1 = {};
        k512_mac2_t<SCHED_DSREAD>(afr, brow0, brow1, acc0, acc1);

        int nb0 = (stage0 + st) * 32;
        float bv0 = bias[nb0 + lo];
        float bv1 = bias[nb0 + 16 + lo];
        #pragma unroll
        for (int r = 0; r < 8; ++r) {
            size_t orow = (size_t)(mbase + hi * 8 + r) * N + nb0;
            out[orow + lo]      = acc0[r] + bv0;
            out[orow + 16 + lo] = acc1[r] + bv1;
        }
        __syncthreads();
    }
}

// ---- GRU: single 512-thread workgroup, h in LDS, 96 WMMA N-tiles / step ----
// dynamic LDS: h_bf16 (16x512x2 = 16KB) + h_f32 (32KB) + gh (16x1536x4 = 96KB)
// W_hh panels are per-wave (no sharing), so B frags stream from L2 with the
// VMEM-read pipelined MAC variant.
__global__ __launch_bounds__(512)
void kGru(const float* __restrict__ xproj, const bf16* __restrict__ Whh,
          const float* __restrict__ bhh, bf16* __restrict__ ctxbf) {
    extern __shared__ char smem[];
    bf16*  hbf = (bf16*)smem;                       // [16][512]
    float* hf  = (float*)(smem + NB * DD * 2);      // [16][512]
    float* gh  = hf + NB * DD;                      // [16][1536]

    int tid = threadIdx.x;
    int lane = tid & 31, wave = tid >> 5;
    int lo = lane & 15, hi = lane >> 4;

    for (int e = tid; e < NB * DD; e += 512) { hf[e] = 0.0f; hbf[e] = (bf16)0.0f; }
    __syncthreads();

    for (int t = 0; t < NL; ++t) {
        // gh = h @ Whh^T   (16 waves x 3 dual-tile stages x 16 K-steps)
        v16bf afr[16];
        const bf16* arow = hbf + lo * DD;
        #pragma unroll
        for (int kc = 0; kc < 16; ++kc) afr[kc] = load_frag_a(arow, hi, kc * 32);

        for (int ti = 0; ti < 3; ++ti) {
            int nb0 = (wave * 6 + ti * 2) * 16;
            const bf16* brow0 = Whh + (size_t)(nb0 + lo) * DD + hi * 16;
            const bf16* brow1 = brow0 + (size_t)16 * DD;
            __builtin_prefetch(brow0 + 32 * DD, 0, 1);
            v8f acc0 = {}, acc1 = {};
            k512_mac2_t<SCHED_VMREAD>(afr, brow0, brow1, acc0, acc1);
            #pragma unroll
            for (int r = 0; r < 8; ++r) {
                gh[(hi * 8 + r) * N3 + nb0 + lo]      = acc0[r];
                gh[(hi * 8 + r) * N3 + nb0 + 16 + lo] = acc1[r];
            }
        }
        __syncthreads();

        // gate math + state update (torch GRU, gate order r,z,n)
        #pragma unroll
        for (int i = 0; i < 16; ++i) {
            int e = i * 512 + tid;                  // 0..8191
            int b = e >> 9, j = e & 511;
            size_t xrow = ((size_t)b * NL + t) * N3;
            float xr = xproj[xrow + j];
            float xz = xproj[xrow + DD + j];
            float xn = xproj[xrow + 2 * DD + j];
            float hr = gh[b * N3 + j]          + bhh[j];
            float hz = gh[b * N3 + DD + j]     + bhh[DD + j];
            float hn = gh[b * N3 + 2 * DD + j] + bhh[2 * DD + j];
            float r = 1.0f / (1.0f + __expf(-(xr + hr)));
            float z = 1.0f / (1.0f + __expf(-(xz + hz)));
            float n = tanhf(xn + r * hn);
            float hnew = (1.0f - z) * n + z * hf[e];
            hf[e] = hnew;
            hbf[e] = (bf16)hnew;
            ctxbf[((size_t)b * NL + t) * DD + j] = (bf16)hnew;
        }
        __syncthreads();
    }
}

// ---- context-prediction loss: one wave per (b,l) pair, per k -----------------
__global__ void kCpLoss(const float* __restrict__ ctxp, const float* __restrict__ feat,
                        float* __restrict__ cpPtr) {
    int k = blockIdx.y + 1;
    int wid = blockIdx.x * (blockDim.x >> 5) + (threadIdx.x >> 5);
    int lane = threadIdx.x & 31;
    int cnt = NB * (NL - k);
    if (wid >= cnt) return;
    int b = wid / (NL - k), l = wid % (NL - k);
    const float* c  = ctxp + ((size_t)b * NL + l) * DD;
    const float* fp = feat + ((size_t)b * NL + l + k) * DD;
    const float* fn = feat + ((size_t)((b + 15) & 15) * NL + l + k) * DD;  // roll(+1, axis=0)
    float sp = 0.0f, sn = 0.0f;
    int j0 = lane * 16;
    #pragma unroll
    for (int j = 0; j < 16; ++j) {
        float cv = c[j0 + j];
        sp += cv * fp[j0 + j];
        sn += cv * fn[j0 + j];
    }
    for (int m = 16; m >= 1; m >>= 1) { sp += __shfl_xor(sp, m, 32); sn += __shfl_xor(sn, m, 32); }
    if (lane == 0) {
        float lk = -logsig(sp) - 0.5f * logsig(-sn);
        atomicAdd(cpPtr, lk / ((float)cnt * 4.0f));
    }
}

__global__ void kFinalize(const float* __restrict__ scal, float* __restrict__ outScal) {
    if (threadIdx.x == 0) {
        float commit = scal[0] / (float)((size_t)NROWS * DD);  // mean (f-q)^2
        float cp = scal[1];
        outScal[0] = cp;
        outScal[1] = commit;          // commitment_loss
        outScal[2] = commit;          // codebook_loss (identical in fwd)
        outScal[3] = cp + commit + 0.5f * commit;
    }
}

// ---- host launcher ----------------------------------------------------------
extern "C" void kernel_launch(void* const* d_in, const int* in_sizes, int n_in,
                              void* d_out, int out_size, void* d_ws, size_t ws_size,
                              hipStream_t stream) {
    (void)in_sizes; (void)n_in; (void)out_size; (void)ws_size;
    const float* feat = (const float*)d_in[0];
    const float* cb   = (const float*)d_in[1];
    const float* Wih  = (const float*)d_in[2];
    const float* Whh  = (const float*)d_in[3];
    const float* bih  = (const float*)d_in[4];
    const float* bhh  = (const float*)d_in[5];
    const float* Wpr  = (const float*)d_in[6];
    const float* bpr  = (const float*)d_in[7];

    float* out_tok  = (float*)d_out;                       // 8192 tokens (as f32)
    float* out_q    = out_tok + NROWS;                     // 8192*512 quantized
    float* out_scal = out_q + (size_t)NROWS * DD;          // 4 scalars

    char* ws = (char*)d_ws;
    size_t off = 0;
    auto carve = [&](size_t bytes) { size_t r = off; off = (off + bytes + 255) & ~(size_t)255; return r; };
    bf16*  featb = (bf16*)(ws + carve((size_t)NROWS * DD * 2));
    bf16*  cbb   = (bf16*)(ws + carve((size_t)NK * DD * 2));
    bf16*  Wihb  = (bf16*)(ws + carve((size_t)N3 * DD * 2));
    bf16*  Whhb  = (bf16*)(ws + carve((size_t)N3 * DD * 2));
    bf16*  Wprb  = (bf16*)(ws + carve((size_t)DD * DD * 2));
    float* cbn   = (float*)(ws + carve((size_t)NK * 4));
    float* pdist = (float*)(ws + carve((size_t)NROWS * SPLITN * 4));
    int*   pidx  = (int*)  (ws + carve((size_t)NROWS * SPLITN * 4));
    int*   idx   = (int*)  (ws + carve((size_t)NROWS * 4));
    bf16*  qb    = (bf16*) (ws + carve((size_t)NROWS * DD * 2));
    float* xproj = (float*)(ws + carve((size_t)NROWS * N3 * 4));
    bf16*  ctxb  = (bf16*) (ws + carve((size_t)NROWS * DD * 2));
    float* ctxp  = (float*)(ws + carve((size_t)NROWS * DD * 4));
    float* scal  = (float*)(ws + carve(64));               // [0]=sse, [1]=cp

    size_t panelLds = (size_t)2 * PANEL32 * sizeof(bf16);  // 66,560 B double-buffered stage

    kZero<<<1, 32, 0, stream>>>(scal, 2);
    kConvert<<<2048, 256, 0, stream>>>(feat, featb, (long)NROWS * DD);
    kConvert<<<2048, 256, 0, stream>>>(cb,   cbb,   (long)NK * DD);
    kConvert<<<768,  256, 0, stream>>>(Wih,  Wihb,  (long)N3 * DD);
    kConvert<<<768,  256, 0, stream>>>(Whh,  Whhb,  (long)N3 * DD);
    kConvert<<<256,  256, 0, stream>>>(Wpr,  Wprb,  (long)DD * DD);
    kRowNorm<<<NK / 8, 256, 0, stream>>>(cb, cbn, NK);

    kArgminGemm<<<dim3(NROWS / 128, SPLITN), 256, panelLds, stream>>>(featb, cbb, cbn, pdist, pidx);
    kCombine<<<NROWS / 256, 256, 0, stream>>>(pdist, pidx, idx, out_tok);
    kGather<<<NROWS, 256, 0, stream>>>(feat, cb, idx, out_q, qb, &scal[0]);

    // x_proj: N=1536 -> 48 stages of 32 cols; grid.y=6 -> 8 stages each
    kGemmBias<<<dim3(NROWS / 128, 6), 256, panelLds, stream>>>(qb, Wihb, bih, xproj, N3, 8);

    size_t gruLds = (size_t)NB * DD * 2 + (size_t)NB * DD * 4 + (size_t)NB * N3 * 4; // 147456
    kGru<<<1, 512, gruLds, stream>>>(xproj, Whhb, bhh, ctxb);

    // projection: N=512 -> 16 stages of 32 cols; grid.y=4 -> 4 stages each
    kGemmBias<<<dim3(NROWS / 128, 4), 256, panelLds, stream>>>(ctxb, Wprb, bpr, ctxp, DD, 4);

    kCpLoss<<<dim3(1024, 4), 256, 0, stream>>>(ctxp, feat, &scal[1]);
    kFinalize<<<1, 32, 0, stream>>>(scal, out_scal);
}
